// UpTransformer_bak_22127671509190
// MI455X (gfx1250) — matrix operand order repaired and checked by reading, not compiled, not run on hardware
//
#include <hip/hip_runtime.h>
#include <hip/hip_bf16.h>
#include <float.h>

// ---------------------------------------------------------------------------
// UpTransformer fused implementation for MI455X (gfx1250, wave32, WMMA).
// All channel-mixing GEMMs run as v_wmma_f32_16x16x32_bf16 (f32 accumulate).
// Every WMMA B-operand is stored N-major (contraction dim contiguous) so each
// lane's fragment is a contiguous 32B run -> global/ds b128 loads, not u16
// chains.  The per-point attention is fully fused (all intermediates in LDS).
// ---------------------------------------------------------------------------

#define B_    8
#define C_    128
#define N_    2048
#define DIM_  64
#define K_    20
#define UP_   2
#define POSH_ 64
#define AH_   256            // DIM*MULT
#define EPS_  1e-5f
#define NUP_  (N_ * UP_)

#define LDP_  72             // padded LDS row (halfs) for 64-wide tiles
#define LDA_  264            // padded LDS row (halfs) for 256-wide tile

typedef __attribute__((ext_vector_type(16))) __bf16 v16bf;
typedef __attribute__((ext_vector_type(8)))  __bf16 v8bf;
typedef __attribute__((ext_vector_type(8)))  float  v8f;

// ---- bf16 helpers (raw ushort storage, RNE convert) -----------------------
static __device__ __forceinline__ unsigned short f2us(float f) {
  unsigned int u = __float_as_uint(f);
  u += 0x7FFFu + ((u >> 16) & 1u);
  return (unsigned short)(u >> 16);
}

// ---- WMMA fragment loaders (contiguous k order, same in A and B) ---------------------------
// A (16x32): lane holds row m = lane&15; halfs h -> K = kb + h,
// kb = (lane&16)?16:0.  Two aligned 16B chunks of one row.
static __device__ __forceinline__ v16bf load_a_frag(
    const unsigned short* __restrict__ W, int ldw, int m0, int k0, int lane) {
  const int kb = (lane & 16) ? 16 : 0;
  const unsigned short* row = W + (long)(m0 + (lane & 15)) * ldw + k0 + kb;
  const v8bf lo = *reinterpret_cast<const v8bf*>(row);
  const v8bf hi = *reinterpret_cast<const v8bf*>(row + 8);
  return __builtin_shufflevector(lo, hi, 0, 1, 2, 3, 4, 5, 6, 7,
                                 8, 9, 10, 11, 12, 13, 14, 15);
}
// B (32x16): lane holds column n = lane&15; halfs h -> K = kb + h,
// kb = (lane&16)?16:0.  With X stored N-major (K contiguous, row stride ld),
// that is one contiguous 32B run (two aligned 16B chunks).
static __device__ __forceinline__ v16bf load_b_frag(
    const unsigned short* __restrict__ X, int ld, int k0, int n0, int lane) {
  const unsigned short* row =
      X + (long)(n0 + (lane & 15)) * ld + k0 + ((lane & 16) ? 16 : 0);
  const v8bf lo = *reinterpret_cast<const v8bf*>(row);
  const v8bf hi = *reinterpret_cast<const v8bf*>(row + 8);
  return __builtin_shufflevector(lo, hi, 0, 1, 2, 3, 4, 5, 6, 7,
                                 8, 9, 10, 11, 12, 13, 14, 15);
}
static __device__ __forceinline__ v8f wmma_bf16(v16bf a, v16bf b, v8f c) {
  return __builtin_amdgcn_wmma_f32_16x16x32_bf16(
      false, a, false, b, (short)0, c, false, false);
}

// ---------------------------------------------------------------------------
// Prep kernels
// ---------------------------------------------------------------------------
__global__ void k_cvt(unsigned short* __restrict__ dst,
                      const float* __restrict__ src, long n) {
  for (long i = (long)blockIdx.x * blockDim.x + threadIdx.x; i < n;
       i += (long)gridDim.x * blockDim.x)
    dst[i] = f2us(src[i]);
}

// (B,C,N) f32 -> (B,N,C) bf16
__global__ void k_cvt_t(unsigned short* __restrict__ dst,
                        const float* __restrict__ src) {
  const long total = (long)B_ * C_ * N_;
  for (long i = (long)blockIdx.x * blockDim.x + threadIdx.x; i < total;
       i += (long)gridDim.x * blockDim.x) {
    const int c = (int)(i % C_);
    const int n = (int)((i / C_) % N_);
    const int b = (int)(i / ((long)C_ * N_));
    dst[i] = f2us(src[((long)b * C_ + c) * N_ + n]);
  }
}

// key,query (B,C,N) -> Xcat (B,N,2C) bf16
__global__ void k_cat(unsigned short* __restrict__ dst,
                      const float* __restrict__ key,
                      const float* __restrict__ qry) {
  const long total = (long)B_ * N_ * 2 * C_;
  for (long i = (long)blockIdx.x * blockDim.x + threadIdx.x; i < total;
       i += (long)gridDim.x * blockDim.x) {
    const int c = (int)(i % (2 * C_));
    const int n = (int)((i / (2 * C_)) % N_);
    const int b = (int)(i / ((long)2 * C_ * N_));
    const float v = (c < C_) ? key[((long)b * C_ + c) * N_ + n]
                             : qry[((long)b * C_ + (c - C_)) * N_ + n];
    dst[i] = f2us(v);
  }
}

// a_wt (Ci=256, Co=64, Up=2)  ->  dst[r][o][c] bf16 (two 64x256 row-major)
__global__ void k_awt(unsigned short* __restrict__ dst,
                      const float* __restrict__ awt) {
  const long total = (long)UP_ * DIM_ * AH_;
  for (long i = (long)blockIdx.x * blockDim.x + threadIdx.x; i < total;
       i += (long)gridDim.x * blockDim.x) {
    const int c = (int)(i % AH_);
    const int o = (int)((i / AH_) % DIM_);
    const int r = (int)(i / ((long)AH_ * DIM_));
    dst[i] = f2us(awt[((long)c * DIM_ + o) * UP_ + r]);
  }
}

__global__ void k_bnfold(float* __restrict__ s, float* __restrict__ sh,
                         const float* __restrict__ g, const float* __restrict__ be,
                         const float* __restrict__ m, const float* __restrict__ v,
                         int n) {
  const int i = blockIdx.x * blockDim.x + threadIdx.x;
  if (i < n) {
    const float sc = g[i] * rsqrtf(v[i] + EPS_);
    s[i] = sc;
    sh[i] = be[i] - m[i] * sc;
  }
}

// ---------------------------------------------------------------------------
// Generic batched WMMA GEMM: Y[b] = act(W(MxKd) @ X[b] + bias (+ addT))
//   X    : N-major bf16, row stride ldx halfs, per-batch stride xbatch halfs
//   addT : optional f32, N-major (Ncols x M), batch stride Ncols*M
//   outT : optional f32, N-major;  outB : optional bf16, N-major
// One wave owns a 16x64 strip (A fragment reused across 4 B fragments);
// all epilogue stores are 8-element contiguous vector stores.
// ---------------------------------------------------------------------------
__global__ __launch_bounds__(128) void k_gemm(
    const unsigned short* __restrict__ W,
    const unsigned short* __restrict__ X, long xbatch, int ldx,
    const float* __restrict__ bias,
    const float* __restrict__ addT,
    float* __restrict__ outT,
    unsigned short* __restrict__ outB,
    int M, int Kd, int Ncols, int relu) {
  const int lane = threadIdx.x & 31;
  const int wv   = threadIdx.x >> 5;
  const int mt   = blockIdx.y * 4 + wv;
  if (mt * 16 >= M) return;                       // wave-uniform
  const int b  = blockIdx.z;
  const int n0 = blockIdx.x * 64;
  const unsigned short* Xb = X + (long)b * xbatch;

  v8f acc[4] = {};
  for (int k0 = 0; k0 < Kd; k0 += 32) {
    const v16bf a = load_a_frag(W, Kd, mt * 16, k0, lane);
#pragma unroll
    for (int t = 0; t < 4; ++t) {
      const v16bf bf = load_b_frag(Xb, ldx, k0, n0 + 16 * t, lane);
      acc[t] = wmma_bf16(a, bf, acc[t]);
    }
  }
  const int mbase = mt * 16 + ((lane & 16) ? 8 : 0);
  const v8f bv = *reinterpret_cast<const v8f*>(bias + mbase);
#pragma unroll
  for (int t = 0; t < 4; ++t) {
    const int n = n0 + 16 * t + (lane & 15);
    const long rowoff = ((long)b * Ncols + n) * M + mbase;
    v8f addv = {};
    if (addT) addv = *reinterpret_cast<const v8f*>(addT + rowoff);
    alignas(32) float buf[8];
    alignas(16) unsigned short hb[8];
#pragma unroll
    for (int r = 0; r < 8; ++r) {
      float v = acc[t][r] + bv[r] + addv[r];
      if (relu) v = fmaxf(v, 0.f);
      buf[r] = v;
      hb[r]  = f2us(v);
    }
    if (outT)
      *reinterpret_cast<v8f*>(outT + rowoff) =
          *reinterpret_cast<const v8f*>(buf);
    if (outB)
      *reinterpret_cast<v8bf*>(outB + rowoff) =
          *reinterpret_cast<const v8bf*>(hb);
  }
}

// ---------------------------------------------------------------------------
// Final GEMM with swapped operand roles so d_out (C-major, n contiguous)
// gets contiguous vector stores:  out^T(4096x128) = agg(4096x64) @ e_w^T.
//   agg  : (B, NUP, 64) bf16 row-major  -> used as the A matrix
//   e_w  : (128, 64) bf16 row-major     -> already e_w^T in N-major form
//   out[b][ch][pt] = acc + e_b[ch] + value[b][pt/2][ch]   (valF is N-major)
// ---------------------------------------------------------------------------
__global__ __launch_bounds__(128) void k_gemm_final(
    const unsigned short* __restrict__ aggB,
    const unsigned short* __restrict__ ewB,
    const float* __restrict__ eb,
    const float* __restrict__ valF,       // (B, N, C) f32
    float* __restrict__ out) {
  const int lane = threadIdx.x & 31;
  const int wv   = threadIdx.x >> 5;
  const int mt   = blockIdx.y * 4 + wv;   // point tile, 0..255
  const int n0   = blockIdx.x * 64;       // out-channel tile base
  const int b    = blockIdx.z;
  const unsigned short* Ab = aggB + (long)b * NUP_ * DIM_;

  v8f acc[4] = {};
#pragma unroll
  for (int ks = 0; ks < 2; ++ks) {
    const v16bf a = load_a_frag(Ab, DIM_, mt * 16, ks * 32, lane);
#pragma unroll
    for (int t = 0; t < 4; ++t) {
      const v16bf bf = load_b_frag(ewB, DIM_, ks * 32, n0 + 16 * t, lane);
      acc[t] = wmma_bf16(a, bf, acc[t]);
    }
  }
  const int mbase = mt * 16 + ((lane & 16) ? 8 : 0);   // point base (even)
#pragma unroll
  for (int t = 0; t < 4; ++t) {
    const int nch = n0 + 16 * t + (lane & 15);
    const float bia = eb[nch];
    alignas(32) float buf[8];
#pragma unroll
    for (int r = 0; r < 8; ++r) {
      const int mp = mbase + r;
      buf[r] = acc[t][r] + bia + valF[((long)b * N_ + (mp >> 1)) * C_ + nch];
    }
    *reinterpret_cast<v8f*>(out + ((long)b * C_ + nch) * NUP_ + mbase) =
        *reinterpret_cast<const v8f*>(buf);
  }
}

// ---------------------------------------------------------------------------
// kNN: one 256-thread block per (b,n); distances in LDS, 20 argmin rounds.
// ---------------------------------------------------------------------------
__global__ __launch_bounds__(256) void k_knn(const float* __restrict__ pos,
                                             int* __restrict__ idx) {
  __shared__ float dist[N_];
  __shared__ float rv[256];
  __shared__ int   ri[256];
  const int tid = threadIdx.x;
  const int b = blockIdx.x / N_;
  const int n = blockIdx.x % N_;
  const float px = pos[((long)b * 3 + 0) * N_ + n];
  const float py = pos[((long)b * 3 + 1) * N_ + n];
  const float pz = pos[((long)b * 3 + 2) * N_ + n];
  for (int m = tid; m < N_; m += 256) {
    const float dx = px - pos[((long)b * 3 + 0) * N_ + m];
    const float dy = py - pos[((long)b * 3 + 1) * N_ + m];
    const float dz = pz - pos[((long)b * 3 + 2) * N_ + m];
    dist[m] = dx * dx + dy * dy + dz * dz;
  }
  __syncthreads();
  for (int j = 0; j < K_; ++j) {
    float bv = FLT_MAX; int bi = 0;
    for (int m = tid; m < N_; m += 256) {
      const float d = dist[m];
      if (d < bv) { bv = d; bi = m; }
    }
    rv[tid] = bv; ri[tid] = bi;
    __syncthreads();
    for (int s = 128; s > 0; s >>= 1) {
      if (tid < s && rv[tid + s] < rv[tid]) { rv[tid] = rv[tid + s]; ri[tid] = ri[tid + s]; }
      __syncthreads();
    }
    if (tid == 0) {
      idx[((long)b * N_ + n) * K_ + j] = ri[0];
      dist[ri[0]] = FLT_MAX;
    }
    __syncthreads();
  }
}

// ---------------------------------------------------------------------------
// Fused per-point attention. One 128-thread block (4 waves) per (b,n).
// All LDS matrix tiles are neighbor-major (channel contiguous) -> b128 loads.
// ---------------------------------------------------------------------------
__global__ __launch_bounds__(128) void k_fused_attn(
    const float* __restrict__ qT, const float* __restrict__ kT,
    const float* __restrict__ vT, const float* __restrict__ ufT,
    const float* __restrict__ pos, const int* __restrict__ idx,
    const float* __restrict__ p_w1, const float* __restrict__ p_b1,
    const float* __restrict__ p_s, const float* __restrict__ p_sh,
    const unsigned short* __restrict__ p_w2b, const float* __restrict__ p_b2,
    const unsigned short* __restrict__ a_w1b, const float* __restrict__ a_b1,
    const float* __restrict__ a_s, const float* __restrict__ a_sh,
    const unsigned short* __restrict__ a_wtT, const float* __restrict__ a_bt,
    unsigned short* __restrict__ aggB) {
  __shared__ int   l_idx[K_];
  __shared__ float l_pr[3][K_];
  __shared__ alignas(16) float          l_val[K_][DIM_];     // val, f32
  __shared__ alignas(16) unsigned short l_S[32][LDP_];       // S, bf16
  __shared__ alignas(16) unsigned short l_pe1[32][LDP_];     // pos hidden
  __shared__ alignas(32) float          l_pe[32][DIM_];      // pe, f32
  __shared__ alignas(16) unsigned short l_A[32][LDA_];       // attn hidden
  __shared__ alignas(32) float          l_T[2][32][DIM_];    // logits

  const int tid  = threadIdx.x;
  const int lane = tid & 31;
  const int wv   = tid >> 5;
  const int b = blockIdx.x / N_;
  const int n = blockIdx.x % N_;

  if (tid < K_) l_idx[tid] = idx[((long)b * N_ + n) * K_ + tid];
  __syncthreads();

  if (tid < 3 * K_) {
    const int d = tid / K_, k = tid % K_;
    l_pr[d][k] = pos[((long)b * 3 + d) * N_ + n] -
                 pos[((long)b * 3 + d) * N_ + l_idx[k]];
  }
  __syncthreads();

  // pe1 = relu(bn(p_w1 @ pos_rel + p_b1)) -> bf16, neighbor-major, zero pad
  for (int e = tid; e < POSH_ * 32; e += 128) {
    const int ch = e & 63, k = e >> 6;
    unsigned short o = 0;
    if (k < K_) {
      float v = p_w1[ch * 3 + 0] * l_pr[0][k] + p_w1[ch * 3 + 1] * l_pr[1][k] +
                p_w1[ch * 3 + 2] * l_pr[2][k] + p_b1[ch];
      v = fmaxf(v * p_s[ch] + p_sh[ch], 0.f);
      o = f2us(v);
    }
    l_pe1[k][ch] = o;
  }
  __syncthreads();

  // pe = p_w2 @ pe1 + p_b2   (M=64, Kd=64, 32 cols): 1 m-tile per wave
  {
    const int mt = wv;
#pragma unroll
    for (int nt = 0; nt < 2; ++nt) {
      v8f acc = {};
#pragma unroll
      for (int ks = 0; ks < 2; ++ks) {
        const v16bf a  = load_a_frag(p_w2b, POSH_, mt * 16, ks * 32, lane);
        const v16bf bb = load_b_frag(&l_pe1[0][0], LDP_, ks * 32, nt * 16, lane);
        acc = wmma_bf16(a, bb, acc);
      }
      const int c = nt * 16 + (lane & 15);
      const int mbase = mt * 16 + ((lane & 16) ? 8 : 0);
      if (c < K_) {
        alignas(32) float buf[8];
#pragma unroll
        for (int r = 0; r < 8; ++r) buf[r] = acc[r] + p_b2[mbase + r];
        *reinterpret_cast<v8f*>(&l_pe[c][mbase]) =
            *reinterpret_cast<const v8f*>(buf);
      }
    }
  }
  __syncthreads();

  // S = q - k_g + pe + uf_rel (bf16); val = v_g + pe + uf_rel (f32)
  // consecutive lanes walk the channel dim -> coalesced 256B gathers
  for (int e = tid; e < DIM_ * 32; e += 128) {
    const int ch = e & 63, k = e >> 6;
    if (k < K_) {
      const int g = l_idx[k];
      const long pbase = ((long)b * N_ + n) * DIM_ + ch;
      const long gbase = ((long)b * N_ + g) * DIM_ + ch;
      const float pe  = l_pe[k][ch];
      const float ufr = ufT[pbase] - ufT[gbase];
      l_val[k][ch] = vT[gbase] + pe + ufr;
      l_S[k][ch]   = f2us(qT[pbase] - kT[gbase] + pe + ufr);
    } else {
      l_S[k][ch] = 0;
    }
  }
  __syncthreads();

  // A = relu(bn(a_w1 @ S + a_b1))   (M=256, Kd=64): 4 m-tiles per wave
#pragma unroll
  for (int mi = 0; mi < 4; ++mi) {
    const int mt = wv * 4 + mi;
#pragma unroll
    for (int nt = 0; nt < 2; ++nt) {
      v8f acc = {};
#pragma unroll
      for (int ks = 0; ks < 2; ++ks) {
        const v16bf a  = load_a_frag(a_w1b, DIM_, mt * 16, ks * 32, lane);
        const v16bf bb = load_b_frag(&l_S[0][0], LDP_, ks * 32, nt * 16, lane);
        acc = wmma_bf16(a, bb, acc);
      }
      const int c = nt * 16 + (lane & 15);
      const int mbase = mt * 16 + ((lane & 16) ? 8 : 0);
      alignas(16) unsigned short hb[8];
#pragma unroll
      for (int r = 0; r < 8; ++r) {
        const int m = mbase + r;
        float v = acc[r] + a_b1[m];
        v = fmaxf(v * a_s[m] + a_sh[m], 0.f);
        hb[r] = f2us(v);
      }
      *reinterpret_cast<v8bf*>(&l_A[c][mbase]) =
          *reinterpret_cast<const v8bf*>(hb);
    }
  }
  __syncthreads();

  // logits: T[r] = a_wtT[r](64x256) @ A + a_bt   (Kd=256)
#pragma unroll
  for (int r2 = 0; r2 < 2; ++r2) {
    const unsigned short* Wt = a_wtT + (long)r2 * DIM_ * AH_;
    const int mt = wv;
#pragma unroll
    for (int nt = 0; nt < 2; ++nt) {
      v8f acc = {};
#pragma unroll
      for (int ks = 0; ks < 8; ++ks) {
        const v16bf a  = load_a_frag(Wt, AH_, mt * 16, ks * 32, lane);
        const v16bf bb = load_b_frag(&l_A[0][0], LDA_, ks * 32, nt * 16, lane);
        acc = wmma_bf16(a, bb, acc);
      }
      const int c = nt * 16 + (lane & 15);
      const int mbase = mt * 16 + ((lane & 16) ? 8 : 0);
      alignas(32) float buf[8];
#pragma unroll
      for (int r = 0; r < 8; ++r) buf[r] = acc[r] + a_bt[mbase + r];
      *reinterpret_cast<v8f*>(&l_T[r2][c][mbase]) =
          *reinterpret_cast<const v8f*>(buf);
    }
  }
  __syncthreads();

  // softmax over K fused with weighted sum: one thread per (r, ch);
  // lanes read 64 consecutive floats per k -> conflict-free
  {
    const int r2 = tid >> 6;
    const int ch = tid & 63;
    float mx = -FLT_MAX;
#pragma unroll
    for (int k = 0; k < K_; ++k) mx = fmaxf(mx, l_T[r2][k][ch]);
    float s = 0.f, acc = 0.f;
#pragma unroll
    for (int k = 0; k < K_; ++k) {
      const float e = __expf(l_T[r2][k][ch] - mx);
      s   += e;
      acc += e * l_val[k][ch];
    }
    // agg stored (B, NUP, 64) row-major = A-operand layout for final GEMM
    aggB[((long)b * NUP_ + 2 * n + r2) * DIM_ + ch] = f2us(acc / s);
  }
}

// ---------------------------------------------------------------------------
// Host side
// ---------------------------------------------------------------------------
enum {
  I_POS = 0, I_KEY, I_QUERY, I_UPFEAT,
  P_VW1, P_VB1, P_VW2, P_VB2, P_VWS, P_VBS,
  P_KW, P_KB, P_QW, P_QB, P_VVW, P_VVB, P_UW, P_UB,
  P_PW1, P_PB1, P_PG, P_PBE, P_PM, P_PV, P_PW2, P_PB2,
  P_AW1, P_AB1, P_AG, P_ABE, P_AM, P_AV, P_AWT, P_ABT,
  P_EW, P_EB
};

extern "C" void kernel_launch(void* const* d_in, const int* in_sizes, int n_in,
                              void* d_out, int out_size, void* d_ws, size_t ws_size,
                              hipStream_t stream) {
  (void)in_sizes; (void)n_in; (void)out_size; (void)ws_size;
  auto F = [&](int i) { return (const float*)d_in[i]; };

  char* ws = (char*)d_ws;
  size_t off = 0;
  auto carve = [&](size_t bytes) -> void* {
    void* p = ws + off;
    off = (off + bytes + 255) & ~(size_t)255;
    return p;
  };

  // activations (all WMMA inputs N-major, contraction contiguous)
  unsigned short* wXcat = (unsigned short*)carve((size_t)B_ * N_ * 2 * C_ * 2); // (B,N,256)
  unsigned short* wUpB  = (unsigned short*)carve((size_t)B_ * N_ * C_ * 2);     // (B,N,128)
  unsigned short* wH    = (unsigned short*)carve((size_t)B_ * N_ * C_ * 2);     // (B,N,128)
  float*          wTmp  = (float*)carve((size_t)B_ * N_ * C_ * 4);              // (B,N,128)
  float*          wValF = (float*)carve((size_t)B_ * N_ * C_ * 4);              // (B,N,128)
  unsigned short* wValB = (unsigned short*)carve((size_t)B_ * N_ * C_ * 2);     // (B,N,128)
  float*          wKT   = (float*)carve((size_t)B_ * N_ * DIM_ * 4);            // (B,N,64)
  float*          wQT   = (float*)carve((size_t)B_ * N_ * DIM_ * 4);
  float*          wVT   = (float*)carve((size_t)B_ * N_ * DIM_ * 4);
  float*          wUT   = (float*)carve((size_t)B_ * N_ * DIM_ * 4);
  int*            wIdx  = (int*)carve((size_t)B_ * N_ * K_ * 4);
  unsigned short* wAggB = (unsigned short*)carve((size_t)B_ * NUP_ * DIM_ * 2); // (B,NUP,64)
  // bf16 weights (row-major, K contiguous)
  unsigned short* wW1   = (unsigned short*)carve((size_t)C_ * 2 * C_ * 2);
  unsigned short* wW2   = (unsigned short*)carve((size_t)C_ * C_ * 2);
  unsigned short* wWs   = (unsigned short*)carve((size_t)C_ * 2 * C_ * 2);
  unsigned short* wKw   = (unsigned short*)carve((size_t)DIM_ * C_ * 2);
  unsigned short* wQw   = (unsigned short*)carve((size_t)DIM_ * C_ * 2);
  unsigned short* wVw   = (unsigned short*)carve((size_t)DIM_ * C_ * 2);
  unsigned short* wUw   = (unsigned short*)carve((size_t)DIM_ * C_ * 2);
  unsigned short* wPw2  = (unsigned short*)carve((size_t)DIM_ * POSH_ * 2);
  unsigned short* wAw1  = (unsigned short*)carve((size_t)AH_ * DIM_ * 2);
  unsigned short* wAwtT = (unsigned short*)carve((size_t)UP_ * DIM_ * AH_ * 2);
  unsigned short* wEw   = (unsigned short*)carve((size_t)C_ * DIM_ * 2);
  // folded BN
  float* wPs  = (float*)carve(POSH_ * 4);
  float* wPsh = (float*)carve(POSH_ * 4);
  float* wAs  = (float*)carve(AH_ * 4);
  float* wAsh = (float*)carve(AH_ * 4);

  // ---- prep ---------------------------------------------------------------
  k_cvt<<<64, 256, 0, stream>>>(wW1,  F(P_VW1), (long)C_ * 2 * C_);
  k_cvt<<<64, 256, 0, stream>>>(wW2,  F(P_VW2), (long)C_ * C_);
  k_cvt<<<64, 256, 0, stream>>>(wWs,  F(P_VWS), (long)C_ * 2 * C_);
  k_cvt<<<32, 256, 0, stream>>>(wKw,  F(P_KW),  (long)DIM_ * C_);
  k_cvt<<<32, 256, 0, stream>>>(wQw,  F(P_QW),  (long)DIM_ * C_);
  k_cvt<<<32, 256, 0, stream>>>(wVw,  F(P_VVW), (long)DIM_ * C_);
  k_cvt<<<32, 256, 0, stream>>>(wUw,  F(P_UW),  (long)DIM_ * C_);
  k_cvt<<<16, 256, 0, stream>>>(wPw2, F(P_PW2), (long)DIM_ * POSH_);
  k_cvt<<<64, 256, 0, stream>>>(wAw1, F(P_AW1), (long)AH_ * DIM_);
  k_cvt<<<32, 256, 0, stream>>>(wEw,  F(P_EW),  (long)C_ * DIM_);
  k_awt<<<64, 256, 0, stream>>>(wAwtT, F(P_AWT));
  k_cat<<<512, 256, 0, stream>>>(wXcat, F(I_KEY), F(I_QUERY));
  k_cvt_t<<<512, 256, 0, stream>>>(wUpB, F(I_UPFEAT));
  k_bnfold<<<1, 64, 0, stream>>>(wPs, wPsh, F(P_PG), F(P_PBE), F(P_PM), F(P_PV), POSH_);
  k_bnfold<<<1, 256, 0, stream>>>(wAs, wAsh, F(P_AG), F(P_ABE), F(P_AM), F(P_AV), AH_);

  // ---- kNN ----------------------------------------------------------------
  k_knn<<<B_ * N_, 256, 0, stream>>>(F(I_POS), wIdx);

  // ---- value = MLP_Res(cat(key,query)) ------------------------------------
  // h = relu(W1 @ X + b1)           (bf16 N-major out)
  k_gemm<<<dim3(N_ / 64, 2, B_), 128, 0, stream>>>(
      wW1, wXcat, (long)N_ * 2 * C_, 2 * C_, F(P_VB1),
      nullptr, nullptr, wH, C_, 2 * C_, N_, 1);
  // tmp = Ws @ X + bs               (f32 N-major out)
  k_gemm<<<dim3(N_ / 64, 2, B_), 128, 0, stream>>>(
      wWs, wXcat, (long)N_ * 2 * C_, 2 * C_, F(P_VBS),
      nullptr, wTmp, nullptr, C_, 2 * C_, N_, 0);
  // value = W2 @ h + b2 + tmp       (f32 for residual, bf16 for v-projection)
  k_gemm<<<dim3(N_ / 64, 2, B_), 128, 0, stream>>>(
      wW2, wH, (long)N_ * C_, C_, F(P_VB2),
      wTmp, wValF, wValB, C_, C_, N_, 0);

  // ---- DIM=64 projections, stored point-major (B,N,64) --------------------
  k_gemm<<<dim3(N_ / 64, 1, B_), 128, 0, stream>>>(
      wKw, wXcat, (long)N_ * 2 * C_, 2 * C_, F(P_KB),
      nullptr, wKT, nullptr, DIM_, C_, N_, 0);
  k_gemm<<<dim3(N_ / 64, 1, B_), 128, 0, stream>>>(
      wQw, wXcat + C_, (long)N_ * 2 * C_, 2 * C_, F(P_QB),
      nullptr, wQT, nullptr, DIM_, C_, N_, 0);
  k_gemm<<<dim3(N_ / 64, 1, B_), 128, 0, stream>>>(
      wVw, wValB, (long)N_ * C_, C_, F(P_VVB),
      nullptr, wVT, nullptr, DIM_, C_, N_, 0);
  k_gemm<<<dim3(N_ / 64, 1, B_), 128, 0, stream>>>(
      wUw, wUpB, (long)N_ * C_, C_, F(P_UB),
      nullptr, wUT, nullptr, DIM_, C_, N_, 0);

  // ---- fused per-point attention -----------------------------------------
  k_fused_attn<<<B_ * N_, 128, 0, stream>>>(
      wQT, wKT, wVT, wUT, F(I_POS), wIdx,
      F(P_PW1), F(P_PB1), wPs, wPsh, wPw2, F(P_PB2),
      wAw1, F(P_AB1), wAs, wAsh, wAwtT, F(P_ABT), wAggB);

  // ---- out = e_w @ agg + e_b + repeat(value) ------------------------------
  k_gemm_final<<<dim3(2, 64, B_), 128, 0, stream>>>(
      wAggB, wEw, F(P_EB), wValF, (float*)d_out);
}
